// Encoder_38757784879702
// MI455X (gfx1250) — matrix operand order repaired and checked by reading, not compiled, not run on hardware
//
#include <hip/hip_runtime.h>
#include <stdint.h>

typedef __attribute__((ext_vector_type(16))) __bf16 v16bf;
typedef __attribute__((ext_vector_type(8)))  float  v8f;
typedef __attribute__((ext_vector_type(4)))  unsigned int v4u;
typedef __attribute__((ext_vector_type(8)))  int v8i;
typedef __attribute__((ext_vector_type(4)))  int v4i;

static constexpr int NN  = 100000;   // nodes
static constexpr int NE  = 800000;   // edges
static constexpr int DIN = 128;
static constexpr int DH  = 512;
static constexpr int KC  = 64;       // K-chunk staged per TDM op
static constexpr int RS  = KC + 4;   // padded LDS row stride (floats): 4-dword pad / 64-dword row

// ---------------------------------------------------------------- utilities
__global__ void k_zero_f(float* __restrict__ p, int n) {
  int i = blockIdx.x * blockDim.x + threadIdx.x;
  int st = gridDim.x * blockDim.x;
  for (; i < n; i += st) p[i] = 0.0f;
}

__global__ void k_cvt_bf16(const float* __restrict__ s, __bf16* __restrict__ d, int n) {
  int i = blockIdx.x * blockDim.x + threadIdx.x;
  int st = gridDim.x * blockDim.x;
  for (; i < n; i += st) d[i] = (__bf16)s[i];
}

__global__ void k_degree(const int* __restrict__ dstI, float* __restrict__ deg, int nE) {
  int i = blockIdx.x * blockDim.x + threadIdx.x;
  int st = gridDim.x * blockDim.x;
  for (; i < nE; i += st) atomicAdd(&deg[dstI[i]], 1.0f);
}

__global__ void k_invdeg(float* __restrict__ deg, int n) {
  int i = blockIdx.x * blockDim.x + threadIdx.x;
  if (i < n) deg[i] = 1.0f / fmaxf(deg[i], 1.0f);
}

// Edge scatter-add: each edge handled by (D/4) lanes, one float4 per lane.
// lgc = log2(D/4): 5 for D=128, 7 for D=512.
__global__ void k_scatter(const float* __restrict__ feat, const int* __restrict__ srcI,
                          const int* __restrict__ dstI, float* __restrict__ agg,
                          int nE, int D, int lgc) {
  int e = blockIdx.x * (blockDim.x >> lgc) + (threadIdx.x >> lgc);
  if (e >= nE) return;
  int c = (threadIdx.x & ((1 << lgc) - 1)) << 2;
  int s = srcI[e], d = dstI[e];
  float4 v = *(const float4*)(feat + (size_t)s * D + c);
  float* o = agg + (size_t)d * D + c;
  atomicAdd(o + 0, v.x);
  atomicAdd(o + 1, v.y);
  atomicAdd(o + 2, v.z);
  atomicAdd(o + 3, v.w);
}

// ---------------------------------------------------- TDM async tile staging
// Load a 32 x KC fp32 tile (row stride K elements) from global into LDS.
// D# per CDNA5 ISA ch.8: group0 = {count, lds_addr, global_addr, type=2},
// group1 = {data_size=4B | pad_enable | pad_interval(64dw) | pad_amount(4dw),
//           tensor_dim0=KC, tensor_dim1=32, tile_dim0=KC, tile_dim1=32,
//           tensor_dim0_stride=K}. Tracked by TENSORcnt.
__device__ __forceinline__ void tdm_load_tile(const float* __restrict__ g,
                                              unsigned ldsByteOff, int K) {
  uint64_t ga = (uint64_t)(uintptr_t)g;
  v4u g0;
  g0[0] = 1u;                                                // count=1
  g0[1] = ldsByteOff;                                        // lds_addr
  g0[2] = (unsigned)(ga & 0xffffffffu);                      // global_addr[31:0]
  g0[3] = (unsigned)((ga >> 32) & 0x1ffffffu) | (2u << 30);  // addr[56:32] | type=2
  v8i g1;
  g1[0] = (int)((2u << 16) | (1u << 20) | (5u << 22) | (3u << 25));
  g1[1] = (int)(((unsigned)KC & 0xffffu) << 16);  // tensor_dim0[15:0]
  g1[2] = (int)(32u << 16);                       // tensor_dim1[15:0] = 32 rows
  g1[3] = (int)((unsigned)KC << 16);              // tile_dim0 = KC
  g1[4] = 32;                                     // tile_dim1 = 32
  g1[5] = K;                                      // tensor_dim0_stride
  g1[6] = 0;
  g1[7] = 0;
  v4i z4 = {0, 0, 0, 0};
  v8i z8 = {0, 0, 0, 0, 0, 0, 0, 0};
  __builtin_amdgcn_tensor_load_to_lds(g0, g1, z4, z4, z8, 0);  // 6-arg form (clang-23)
}

// ------------------------------------------------------------- WMMA fragments
// A fragment from LDS fp32 row (16x32 bf16): lane l -> row (l&15);
// lanes<16 take K k..k+15; lanes>=16 take K k+16..k+31.
__device__ __forceinline__ v16bf cvt_a_frag(const float* __restrict__ p, int half_, float s) {
  const float* ap = p + half_ * 16;
  float4 p0 = *(const float4*)(ap + 0);
  float4 p1 = *(const float4*)(ap + 4);
  float4 p2 = *(const float4*)(ap + 8);
  float4 p3 = *(const float4*)(ap + 12);
  v16bf f;
  f[0]  = (__bf16)(p0.x * s); f[1]  = (__bf16)(p0.y * s);
  f[2]  = (__bf16)(p0.z * s); f[3]  = (__bf16)(p0.w * s);
  f[4]  = (__bf16)(p1.x * s); f[5]  = (__bf16)(p1.y * s);
  f[6]  = (__bf16)(p1.z * s); f[7]  = (__bf16)(p1.w * s);
  f[8]  = (__bf16)(p2.x * s); f[9]  = (__bf16)(p2.y * s);
  f[10] = (__bf16)(p2.z * s); f[11] = (__bf16)(p2.w * s);
  f[12] = (__bf16)(p3.x * s); f[13] = (__bf16)(p3.y * s);
  f[14] = (__bf16)(p3.z * s); f[15] = (__bf16)(p3.w * s);
  return f;
}

// B fragment (32x16 bf16): lane l -> output channel (l&15); lanes<16 K k..k+15,
// lanes>=16 K k+16..k+31. W stored [out_ch][k] row-major -> one 32B vector load.
__device__ __forceinline__ v16bf load_b_frag(const __bf16* __restrict__ w, int K, int col,
                                             int k0, int half_) {
  return *(const v16bf*)(w + (size_t)col * K + k0 + half_ * 16);
}

// One GEMM accumulation pass: acc[rt][t] += (a * rowScale) @ w^T over K,
// with TDM double-buffered A staging in LDS shared by all 8 waves.
__device__ __forceinline__ void gemm_pass(float (*sA)[32 * RS + 8],
                                          const float* __restrict__ a,
                                          const __bf16* __restrict__ w, int K,
                                          int rowBase, int wave, int half_, int lid,
                                          int colBase, float sc0, float sc1,
                                          v8f (&acc)[2][4]) {
  const int nc = K / KC;
  if (wave == 0)
    tdm_load_tile(a + (size_t)rowBase * K, (unsigned)(uintptr_t)&sA[0][0], K);
  for (int c = 0; c < nc; ++c) {
    if (wave == 0) {
      if (c + 1 < nc) {
        tdm_load_tile(a + (size_t)rowBase * K + (size_t)(c + 1) * KC,
                      (unsigned)(uintptr_t)&sA[(c + 1) & 1][0], K);
        __builtin_amdgcn_s_wait_tensorcnt(1);  // chunk c landed (in-order)
      } else {
        __builtin_amdgcn_s_wait_tensorcnt(0);
      }
    }
    __syncthreads();
    const float* buf = &sA[c & 1][0];
#pragma unroll
    for (int ks = 0; ks < KC; ks += 32) {
      v16bf a0f = cvt_a_frag(buf + lid * RS + ks, half_, sc0);
      v16bf a1f = cvt_a_frag(buf + (lid + 16) * RS + ks, half_, sc1);
      const int kg = c * KC + ks;
#pragma unroll
      for (int t = 0; t < 4; ++t) {
        v16bf bf = load_b_frag(w, K, colBase + t * 16 + lid, kg, half_);
        acc[0][t] = __builtin_amdgcn_wmma_f32_16x16x32_bf16(
            false, a0f, false, bf, (short)0, acc[0][t], false, false);
        acc[1][t] = __builtin_amdgcn_wmma_f32_16x16x32_bf16(
            false, a1f, false, bf, (short)0, acc[1][t], false, false);
      }
    }
    __syncthreads();  // everyone done reading buf before it is re-filled
  }
}

// out[n,j] = maybe_prelu( a0[n,:]*s0[n] @ W0[j,:] + a1[n,:] @ W1[j,:] + bias[j]
//                         + e0[n,j] + e1[n,j] )
// Block = 8 wave32s; wave tile 32 rows x 64 cols; block tile 32 x 512 (full width).
__launch_bounds__(256)
__global__ void k_gemm_fused(const float* __restrict__ a0, const __bf16* __restrict__ w0, int K0,
                             const float* __restrict__ scale0,
                             const float* __restrict__ a1, const __bf16* __restrict__ w1, int K1,
                             const float* __restrict__ bias,
                             const float* __restrict__ e0, const float* __restrict__ e1,
                             const float* __restrict__ alpha_p, int do_prelu,
                             float* __restrict__ out) {
  __shared__ float sA[2][32 * RS + 8];  // double-buffered 32 x KC fp32 tile (padded)

  const int lane    = threadIdx.x & 31;
  const int wave    = threadIdx.x >> 5;
  const int half_   = lane >> 4;
  const int lid     = lane & 15;
  const int rowBase = blockIdx.x * 32;
  const int colBase = wave * 64;

  v8f acc[2][4];
  v8f zero = {0.f, 0.f, 0.f, 0.f, 0.f, 0.f, 0.f, 0.f};
#pragma unroll
  for (int rt = 0; rt < 2; ++rt)
#pragma unroll
    for (int t = 0; t < 4; ++t) acc[rt][t] = zero;

  const float s0 = scale0 ? scale0[rowBase + lid] : 1.0f;
  const float s1 = scale0 ? scale0[rowBase + 16 + lid] : 1.0f;

  gemm_pass(sA, a0, w0, K0, rowBase, wave, half_, lid, colBase, s0, s1, acc);
  if (a1)
    gemm_pass(sA, a1, w1, K1, rowBase, wave, half_, lid, colBase, 1.0f, 1.0f, acc);

  // Epilogue: D layout -> VGPR r = row (base + 8*half_ + r), col = lid within tile.
  const float alpha = do_prelu ? alpha_p[0] : 0.0f;
#pragma unroll
  for (int rt = 0; rt < 2; ++rt) {
#pragma unroll
    for (int t = 0; t < 4; ++t) {
      const int col = colBase + t * 16 + lid;
      const float bb = bias[col];
#pragma unroll
      for (int r = 0; r < 8; ++r) {
        const int row = rowBase + rt * 16 + half_ * 8 + r;
        const size_t o = (size_t)row * DH + col;
        float v = acc[rt][t][r] + bb;
        if (e0) v += e0[o];
        if (e1) v += e1[o];
        if (do_prelu && v < 0.0f) v *= alpha;
        out[o] = v;
      }
    }
  }
}

// ---------------------------------------------------------------- launcher
extern "C" void kernel_launch(void* const* d_in, const int* in_sizes, int n_in,
                              void* d_out, int out_size, void* d_ws, size_t ws_size,
                              hipStream_t stream) {
  (void)in_sizes; (void)n_in; (void)out_size; (void)ws_size;

  const float* x     = (const float*)d_in[0];
  const int*   ei    = (const int*)d_in[1];   // [2, NE] int32 (jax default x32)
  const float* Wl1   = (const float*)d_in[2];
  const float* bl1   = (const float*)d_in[3];
  const float* Wr1   = (const float*)d_in[4];
  const float* Wl2   = (const float*)d_in[5];
  const float* bl2   = (const float*)d_in[6];
  const float* Wr2   = (const float*)d_in[7];
  const float* Wl3   = (const float*)d_in[8];
  const float* bl3   = (const float*)d_in[9];
  const float* Wr3   = (const float*)d_in[10];
  const float* Ww    = (const float*)d_in[11];
  const float* bw    = (const float*)d_in[12];
  const float* Ww2   = (const float*)d_in[13];
  const float* bw2   = (const float*)d_in[14];
  const float* alpha = (const float*)d_in[15];

  // workspace carve-up
  char* base = (char*)d_ws;
  size_t off = 0;
  auto alloc = [&](size_t bytes) -> void* {
    void* p = base + off;
    off += (bytes + 255) & ~(size_t)255;
    return p;
  };
  float*  deg  = (float*)alloc((size_t)NN * sizeof(float));       // -> 1/deg
  float*  agg  = (float*)alloc((size_t)NN * DH * sizeof(float));  // mean-agg accum
  float*  h1   = (float*)alloc((size_t)NN * DH * sizeof(float));
  float*  zB   = (float*)alloc((size_t)NN * DH * sizeof(float));  // z2, then z3
  float*  h2   = (float*)alloc((size_t)NN * DH * sizeof(float));
  __bf16* Wl1b = (__bf16*)alloc((size_t)DH * DIN * 2);
  __bf16* Wr1b = (__bf16*)alloc((size_t)DH * DIN * 2);
  __bf16* Wwb  = (__bf16*)alloc((size_t)DH * DIN * 2);
  __bf16* Ww2b = (__bf16*)alloc((size_t)DH * DIN * 2);
  __bf16* Wl2b = (__bf16*)alloc((size_t)DH * DH * 2);
  __bf16* Wr2b = (__bf16*)alloc((size_t)DH * DH * 2);
  __bf16* Wl3b = (__bf16*)alloc((size_t)DH * DH * 2);
  __bf16* Wr3b = (__bf16*)alloc((size_t)DH * DH * 2);

  auto cvt = [&](const float* s, __bf16* d, int n) {
    k_cvt_bf16<<<(n + 255) / 256, 256, 0, stream>>>(s, d, n);
  };
  cvt(Wl1, Wl1b, DH * DIN); cvt(Wr1, Wr1b, DH * DIN);
  cvt(Ww,  Wwb,  DH * DIN); cvt(Ww2, Ww2b, DH * DIN);
  cvt(Wl2, Wl2b, DH * DH);  cvt(Wr2, Wr2b, DH * DH);
  cvt(Wl3, Wl3b, DH * DH);  cvt(Wr3, Wr3b, DH * DH);

  const int* srcI = ei;
  const int* dstI = ei + NE;

  // degree -> 1/max(deg,1)
  k_zero_f<<<(NN + 255) / 256, 256, 0, stream>>>(deg, NN);
  k_degree<<<(NE + 255) / 256, 256, 0, stream>>>(dstI, deg, NE);
  k_invdeg<<<(NN + 255) / 256, 256, 0, stream>>>(deg, NN);

  const dim3 gemmGrid(NN / 32);  // 3125 blocks, each 32 rows x full 512 cols

  // ---- layer 1: agg = segsum(x[src]->dst) (D=128); h1 = PReLU(agg/deg @Wl1^T + bl1 + x@Wr1^T)
  k_zero_f<<<(NN * DIN + 255) / 256, 256, 0, stream>>>(agg, NN * DIN);
  k_scatter<<<NE / 8, 256, 0, stream>>>(x, srcI, dstI, agg, NE, DIN, 5);
  k_gemm_fused<<<gemmGrid, 256, 0, stream>>>(agg, Wl1b, DIN, deg, x, Wr1b, DIN,
                                             bl1, nullptr, nullptr, alpha, 1, h1);

  // ---- z2 = h1 + x@Ww^T + bw
  k_gemm_fused<<<gemmGrid, 256, 0, stream>>>(x, Wwb, DIN, nullptr, nullptr, nullptr, 0,
                                             bw, h1, nullptr, nullptr, 0, zB);

  // ---- layer 2
  k_zero_f<<<(NN * DH + 255) / 256, 256, 0, stream>>>(agg, NN * DH);
  k_scatter<<<NE / 2, 256, 0, stream>>>(zB, srcI, dstI, agg, NE, DH, 7);
  k_gemm_fused<<<gemmGrid, 256, 0, stream>>>(agg, Wl2b, DH, deg, zB, Wr2b, DH,
                                             bl2, nullptr, nullptr, alpha, 1, h2);

  // ---- z3 = h2 + h1 + x@Ww2^T + bw2  (z2 dead, reuse zB)
  k_gemm_fused<<<gemmGrid, 256, 0, stream>>>(x, Ww2b, DIN, nullptr, nullptr, nullptr, 0,
                                             bw2, h2, h1, nullptr, 0, zB);

  // ---- layer 3 -> d_out
  k_zero_f<<<(NN * DH + 255) / 256, 256, 0, stream>>>(agg, NN * DH);
  k_scatter<<<NE / 2, 256, 0, stream>>>(zB, srcI, dstI, agg, NE, DH, 7);
  k_gemm_fused<<<gemmGrid, 256, 0, stream>>>(agg, Wl3b, DH, deg, zB, Wr3b, DH,
                                             bl3, nullptr, nullptr, alpha, 1, (float*)d_out);
}
